// DecoderRNN_6270652253000
// MI455X (gfx1250) — matrix operand order repaired and checked by reading, not compiled, not run on hardware
//
#include <hip/hip_runtime.h>

typedef __attribute__((ext_vector_type(16))) __bf16        v16bf;
typedef __attribute__((ext_vector_type(8)))  float         v8f;
typedef __attribute__((ext_vector_type(4)))  float         f32x4;
typedef __attribute__((ext_vector_type(4)))  unsigned int  u32x4;
typedef __attribute__((ext_vector_type(2)))  unsigned int  u32x2;
typedef __attribute__((ext_vector_type(8)))  int           i32x8;
typedef __attribute__((ext_vector_type(4)))  int           i32x4;

#define HDIM 1024
#define BDIM 64
#define TDIM 50
#define VDIM 32000
#define G4   4096   // 4*H

__device__ __forceinline__ unsigned int f2bf1(float f) {
  unsigned int u = __float_as_uint(f);
  return (u + 0x7FFFu + ((u >> 16) & 1u)) >> 16;   // RNE bf16
}
__device__ __forceinline__ unsigned int packbf(float lo, float hi) {
  return f2bf1(lo) | (f2bf1(hi) << 16);
}

union Frag { u32x4 q[2]; v16bf v; };

// ---------------------------------------------------------------------------
// TDM: DMA a [tile_rows x 32] bf16 tile (tensor row stride 1024 elems) into LDS
// with pad_interval=16 dwords (64B) / pad_amount=4 dwords (16B) -> LDS rows at
// stride 80B (the bank-conflict-free layout the compute waves expect).
// ---------------------------------------------------------------------------
__device__ __forceinline__ void tdm_load_tile(unsigned long long gaddr,
                                              unsigned int lds_byte_off,
                                              unsigned int tensor_rows,
                                              unsigned int tile_rows) {
  u32x4 g0;
  g0[0] = 1u;                                        // count=1, no gather
  g0[1] = lds_byte_off;                              // lds_addr
  g0[2] = (unsigned int)gaddr;                       // global_addr[31:0]
  g0[3] = (unsigned int)(gaddr >> 32) | (2u << 30);  // global_addr[56:32] | type=2
  i32x8 g1;
  g1[0] = (int)((1u << 16)   |                       // data_size = 2 bytes
                (1u << 20)   |                       // pad_enable
                (3u << 22)   |                       // pad_interval: 16 dwords
                (3u << 25));                         // pad_amount: 4 dwords
  g1[1] = (int)(1024u << 16);                        // tensor_dim0 = 1024 (bits 63:48)
  g1[2] = (int)(tensor_rows << 16);                  // dim0 hi=0 | tensor_dim1 lo
  g1[3] = (int)((tensor_rows >> 16) | (32u << 16));  // tensor_dim1 hi | tile_dim0=32
  g1[4] = (int)tile_rows;                            // tile_dim1 | tile_dim2=0
  g1[5] = 1024;                                      // tensor_dim0_stride lo
  g1[6] = 0;                                         // stride hi | dim1_stride lo
  g1[7] = 0;
  i32x4 z = {0, 0, 0, 0};
#if defined(__clang_major__) && __clang_major__ >= 23
  i32x8 z8 = {};
  __builtin_amdgcn_tensor_load_to_lds(g0, g1, z, z, z8, 0);
#else
  __builtin_amdgcn_tensor_load_to_lds(g0, g1, z, z, 0);
#endif
}

// ---------------------------------------------------------------------------
// Big GEMMs: C[M,N] = A[M,1024] * W[N,1024]^T, f32 global staged to bf16 LDS.
// MODE 0: A = relu(emb[token(row)]) fused gather, + b_ih + b_hh, C [M,4096]
// MODE 2: A = hs, + b_out, C permuted: row (t*64+b) -> out row (b*50+t), ldc=V
// ---------------------------------------------------------------------------
template <int MODE>
__global__ __launch_bounds__(256) void gemm_wmma(
    const float* __restrict__ A, const float* __restrict__ W,
    const float* __restrict__ bias0, const float* __restrict__ bias1,
    const int* __restrict__ target, float* __restrict__ C, int M)
{
  __shared__ __align__(16) unsigned short sA[128 * 40];
  __shared__ __align__(16) unsigned short sB[128 * 40];
  __shared__ int sTok[128];

  const int tid   = threadIdx.x;
  const int tileM = blockIdx.y * 128;
  const int tileN = blockIdx.x * 128;

  if (MODE == 0 && tid < 128) {
    int gm = tileM + tid;
    int t = gm >> 6, b = gm & 63;
    sTok[tid] = (t == 0) ? 0 : target[b * TDIM + (t - 1)];
  }

  v8f acc[2][4];
  v8f zero = {};
  #pragma unroll
  for (int i = 0; i < 2; ++i)
    #pragma unroll
    for (int j = 0; j < 4; ++j) acc[i][j] = zero;

  const int lane = tid & 31;
  const int wave = tid >> 5;
  const int wm   = (wave & 3) * 32;
  const int wn   = (wave >> 2) * 64;
  const int half = lane >> 4;
  const int l16  = lane & 15;

  __syncthreads();

  for (int kk = 0; kk < HDIM; kk += 32) {
    #pragma unroll
    for (int i = 0; i < 4; ++i) {
      int chunk = tid + i * 256;
      int r  = chunk >> 3;
      int cc = (chunk & 7) * 4;
      f32x4 av;
      if (MODE == 0) {
        av = *(const f32x4*)(A + (long)sTok[r] * HDIM + kk + cc);
        av.x = fmaxf(av.x, 0.f); av.y = fmaxf(av.y, 0.f);
        av.z = fmaxf(av.z, 0.f); av.w = fmaxf(av.w, 0.f);
      } else {
        av = *(const f32x4*)(A + (long)(tileM + r) * HDIM + kk + cc);
      }
      u32x2 pa; pa.x = packbf(av.x, av.y); pa.y = packbf(av.z, av.w);
      *(u32x2*)&sA[r * 40 + cc] = pa;

      f32x4 bv = *(const f32x4*)(W + (long)(tileN + r) * HDIM + kk + cc);
      u32x2 pb; pb.x = packbf(bv.x, bv.y); pb.y = packbf(bv.z, bv.w);
      *(u32x2*)&sB[r * 40 + cc] = pb;
    }
    __syncthreads();

    Frag a[2], b[4];
    #pragma unroll
    for (int i = 0; i < 2; ++i) {
      const unsigned short* p = &sA[(wm + i * 16 + l16) * 40];
      a[i].q[0] = *(const u32x4*)(p + half * 8);
      a[i].q[1] = *(const u32x4*)(p + 16 + half * 8);
    }
    #pragma unroll
    for (int j = 0; j < 4; ++j) {
      const unsigned short* p = &sB[(wn + j * 16 + l16) * 40];
      b[j].q[0] = *(const u32x4*)(p + half * 8);
      b[j].q[1] = *(const u32x4*)(p + 16 + half * 8);
    }
    #pragma unroll
    for (int i = 0; i < 2; ++i)
      #pragma unroll
      for (int j = 0; j < 4; ++j)
        acc[i][j] = __builtin_amdgcn_wmma_f32_16x16x32_bf16(
            false, a[i].v, false, b[j].v, (short)0, acc[i][j], false, false);
    __syncthreads();
  }

  #pragma unroll
  for (int i = 0; i < 2; ++i) {
    #pragma unroll
    for (int j = 0; j < 4; ++j) {
      int col   = tileN + wn + j * 16 + l16;
      int rbase = tileM + wm + i * 16 + half * 8;
      #pragma unroll
      for (int v = 0; v < 8; ++v) {
        int r = rbase + v;
        float val = acc[i][j][v];
        if (MODE == 0) {
          C[(long)r * G4 + col] = val + bias0[col] + bias1[col];
        } else {
          int t = r >> 6, bb = r & 63;
          C[(long)(bb * TDIM + t) * VDIM + col] = val + bias0[col];
        }
      }
    }
  }
  (void)M;
}

// ---------------------------------------------------------------------------
// Recurrent-step GEMM (M=64, N=4096, K=1024): A=hbf, B=Whh_bf, both already
// bf16 in global. Tile staging via double-buffered TDM DMA, gated by
// s_wait_tensorcnt; hot loop = ds_load fragments + WMMA only.
// Dynamic LDS layout (bytes): A0@0(5120) B0@5120(10240) A1@15360 B1@20480.
// ---------------------------------------------------------------------------
__global__ __launch_bounds__(256) void gemm_rec_tdm(
    const unsigned short* __restrict__ Abf,   // [64,1024] bf16
    const unsigned short* __restrict__ Bbf,   // [4096,1024] bf16
    const float* __restrict__ addend,         // xg + t*64*4096
    float* __restrict__ C)                    // gates [64,4096]
{
  extern __shared__ unsigned short smem[];
  const int tid   = threadIdx.x;
  const int tileN = blockIdx.x * 128;
  const int lane  = tid & 31;
  const int wave  = tid >> 5;
  const int wm    = (wave & 1) * 32;   // 2 waves in M (64 rows)
  const int wn    = (wave >> 1) * 32;  // 4 waves in N (128 cols)
  const int half  = lane >> 4;
  const int l16   = lane & 15;

  v8f acc[2][2];
  v8f zero = {};
  #pragma unroll
  for (int i = 0; i < 2; ++i)
    #pragma unroll
    for (int j = 0; j < 2; ++j) acc[i][j] = zero;

  const unsigned long long aBase = (unsigned long long)Abf;
  const unsigned long long bBase = (unsigned long long)(Bbf + (long)tileN * HDIM);

  if (wave == 0) {
    tdm_load_tile(aBase,      0u,     64u,   64u);
    tdm_load_tile(bBase,      5120u,  4096u, 128u);
    tdm_load_tile(aBase + 64, 15360u, 64u,   64u);
    tdm_load_tile(bBase + 64, 20480u, 4096u, 128u);
  }

  for (int ks = 0; ks < 32; ++ks) {
    if (wave == 0) {
      if (ks == 31) __builtin_amdgcn_s_wait_tensorcnt(0);
      else          __builtin_amdgcn_s_wait_tensorcnt(2);
    }
    __syncthreads();

    const int buf = ks & 1;
    const unsigned short* sA = smem + (buf ? 7680 : 0);      // element offsets
    const unsigned short* sB = smem + (buf ? 10240 : 2560);

    Frag a[2], b[2];
    #pragma unroll
    for (int i = 0; i < 2; ++i) {
      const unsigned short* p = sA + (wm + i * 16 + l16) * 40;
      a[i].q[0] = *(const u32x4*)(p + half * 8);
      a[i].q[1] = *(const u32x4*)(p + 16 + half * 8);
    }
    #pragma unroll
    for (int j = 0; j < 2; ++j) {
      const unsigned short* p = sB + (wn + j * 16 + l16) * 40;
      b[j].q[0] = *(const u32x4*)(p + half * 8);
      b[j].q[1] = *(const u32x4*)(p + 16 + half * 8);
    }
    #pragma unroll
    for (int i = 0; i < 2; ++i)
      #pragma unroll
      for (int j = 0; j < 2; ++j)
        acc[i][j] = __builtin_amdgcn_wmma_f32_16x16x32_bf16(
            false, a[i].v, false, b[j].v, (short)0, acc[i][j], false, false);
    __syncthreads();

    if (wave == 0 && ks + 2 < 32) {
      unsigned long long koff = (unsigned long long)(ks + 2) * 64ull; // bytes
      tdm_load_tile(aBase + koff, buf ? 15360u : 0u,    64u,   64u);
      tdm_load_tile(bBase + koff, buf ? 20480u : 5120u, 4096u, 128u);
    }
  }

  #pragma unroll
  for (int i = 0; i < 2; ++i) {
    #pragma unroll
    for (int j = 0; j < 2; ++j) {
      int col   = tileN + wn + j * 16 + l16;
      int rbase = wm + i * 16 + half * 8;
      #pragma unroll
      for (int v = 0; v < 8; ++v) {
        int r = rbase + v;
        C[(long)r * G4 + col] = acc[i][j][v] + addend[(long)r * G4 + col];
      }
    }
  }
}

// f32 -> bf16 bulk convert
__global__ __launch_bounds__(256) void convert_bf16(
    const float* __restrict__ src, unsigned short* __restrict__ dst, int n)
{
  int i = blockIdx.x * 256 + threadIdx.x;
  if (i < n) dst[i] = (unsigned short)f2bf1(src[i]);
}

// LSTM cell: gates [64,4096] + c -> new c, h (f32), hbf (bf16), hs[t] (f32)
__global__ __launch_bounds__(256) void lstm_cell(
    const float* __restrict__ gates, float* __restrict__ h, float* __restrict__ c,
    unsigned short* __restrict__ hbf, float* __restrict__ hs, int t)
{
  int idx = blockIdx.x * 256 + threadIdx.x;  // 0..65535
  int b = idx >> 10, n = idx & 1023;
  const float* gr = gates + (long)b * G4;
  float gi = gr[n], gf = gr[HDIM + n], gg = gr[2 * HDIM + n], go = gr[3 * HDIM + n];
  float ii = 1.f / (1.f + __expf(-gi));
  float ff = 1.f / (1.f + __expf(-gf));
  float g  = tanhf(gg);
  float oo = 1.f / (1.f + __expf(-go));
  float cn = ff * c[idx] + ii * g;
  float hn = oo * tanhf(cn);
  c[idx]   = cn;
  h[idx]   = hn;
  hbf[idx] = (unsigned short)f2bf1(hn);
  hs[((long)t * BDIM + b) * HDIM + n] = hn;
}

// In-place log_softmax over V=32000, one workgroup per (b,t) row
__global__ __launch_bounds__(256) void logsoftmax_kernel(float* __restrict__ out)
{
  float* row = out + (long)blockIdx.x * VDIM;
  __shared__ float red[256];
  int tid = threadIdx.x;

  float m = -3.4e38f;
  for (int i = tid; i < VDIM; i += 256) m = fmaxf(m, row[i]);
  red[tid] = m; __syncthreads();
  for (int s = 128; s > 0; s >>= 1) {
    if (tid < s) red[tid] = fmaxf(red[tid], red[tid + s]);
    __syncthreads();
  }
  m = red[0]; __syncthreads();

  float sum = 0.f;
  for (int i = tid; i < VDIM; i += 256) sum += __expf(row[i] - m);
  red[tid] = sum; __syncthreads();
  for (int s = 128; s > 0; s >>= 1) {
    if (tid < s) red[tid] += red[tid + s];
    __syncthreads();
  }
  float lse = m + __logf(red[0]);

  for (int i = tid; i < VDIM; i += 256) row[i] -= lse;
}

extern "C" void kernel_launch(void* const* d_in, const int* in_sizes, int n_in,
                              void* d_out, int out_size, void* d_ws, size_t ws_size,
                              hipStream_t stream) {
  (void)in_sizes; (void)n_in; (void)out_size; (void)ws_size;
  const float* enc_h  = (const float*)d_in[1];
  const float* enc_c  = (const float*)d_in[2];
  const int*   target = (const int*)d_in[3];
  const float* emb    = (const float*)d_in[4];
  const float* W_ih   = (const float*)d_in[5];
  const float* W_hh   = (const float*)d_in[6];
  const float* b_ih   = (const float*)d_in[7];
  const float* b_hh   = (const float*)d_in[8];
  const float* W_out  = (const float*)d_in[9];
  const float* b_out  = (const float*)d_in[10];
  float* out = (float*)d_out;

  // workspace: 64 MB f32 region + 8.5 MB bf16 region
  float* xg    = (float*)d_ws;                    // [3200, 4096]
  float* gates = xg    + (long)3200 * G4;         // [64, 4096]
  float* hs    = gates + (long)BDIM * G4;         // [3200, 1024]
  float* h     = hs    + (long)3200 * HDIM;       // [64, 1024]
  float* c     = h     + (long)BDIM * HDIM;       // [64, 1024]
  unsigned short* Whh_bf = (unsigned short*)(c + (long)BDIM * HDIM); // [4096,1024]
  unsigned short* hbf    = Whh_bf + (long)G4 * HDIM;                 // [64,1024]

  hipMemcpyAsync(h, enc_h, (size_t)BDIM * HDIM * sizeof(float), hipMemcpyDeviceToDevice, stream);
  hipMemcpyAsync(c, enc_c, (size_t)BDIM * HDIM * sizeof(float), hipMemcpyDeviceToDevice, stream);

  // one-time bf16 copies for the TDM-fed recurrent GEMM
  convert_bf16<<<dim3(G4 * HDIM / 256), 256, 0, stream>>>(W_hh, Whh_bf, G4 * HDIM);
  convert_bf16<<<dim3(BDIM * HDIM / 256), 256, 0, stream>>>(enc_h, hbf, BDIM * HDIM);

  // x_gates = relu(emb[tokens]) @ W_ih^T + b_ih + b_hh   (M=3200, N=4096)
  gemm_wmma<0><<<dim3(G4 / 128, 3200 / 128), 256, 0, stream>>>(
      emb, W_ih, b_ih, b_hh, target, xg, 3200);

  // sequential scan: gates = x_gates[t] + h @ W_hh^T ; cell update
  for (int t = 0; t < TDIM; ++t) {
    gemm_rec_tdm<<<dim3(G4 / 128), 256, 30720, stream>>>(
        hbf, Whh_bf, xg + (long)t * BDIM * G4, gates);
    lstm_cell<<<dim3(BDIM * HDIM / 256), 256, 0, stream>>>(gates, h, c, hbf, hs, t);
  }

  // logits = hs @ W_out^T + b_out -> d_out permuted to [B,T,V]   (M=3200, N=32000)
  gemm_wmma<2><<<dim3(VDIM / 128, 3200 / 128), 256, 0, stream>>>(
      hs, W_out, b_out, nullptr, nullptr, out, 3200);

  logsoftmax_kernel<<<dim3(BDIM * TDIM), 256, 0, stream>>>(out);

  size_t off = (size_t)BDIM * TDIM * VDIM;
  hipMemcpyAsync(out + off,               h, (size_t)BDIM * HDIM * sizeof(float), hipMemcpyDeviceToDevice, stream);
  hipMemcpyAsync(out + off + BDIM * HDIM, c, (size_t)BDIM * HDIM * sizeof(float), hipMemcpyDeviceToDevice, stream);
}
